// RelativePositionalCrossAttention4D_87952340287647
// MI455X (gfx1250) — matrix operand order repaired and checked by reading, not compiled, not run on hardware
//
#include <hip/hip_runtime.h>
#include <hip/hip_bf16.h>

// Problem constants (from reference)
#define BB 2
#define TT 4
#define NQd 576
#define NKd 576
#define DDim 1024
#define HH 16
#define DH 64
#define MTOK (BB*TT*NQd)          // 4608 token rows
#define NEGBIG (-1.7014118e38f)   // finfo(float32).min / 2

#if __has_builtin(__builtin_amdgcn_global_load_async_to_lds_b128)
#define HAS_ASYNC_LDS 1
#else
#define HAS_ASYNC_LDS 0
#endif

typedef __attribute__((ext_vector_type(16))) _Float16 v16h;
typedef __attribute__((ext_vector_type(8)))  _Float16 v8h;
typedef __attribute__((ext_vector_type(8)))  float    v8f;
typedef __attribute__((ext_vector_type(4)))  int      v4i;

__device__ __forceinline__ v16h cat8(v8h lo, v8h hi) {
  return __builtin_shufflevector(lo, hi, 0,1,2,3,4,5,6,7,8,9,10,11,12,13,14,15);
}
__device__ __forceinline__ v8f wmma_f16(v16h a, v16h b, v8f c) {
  // D = A(16x32,f16) * B(32x16,f16) + C(16x16,f32)
  return __builtin_amdgcn_wmma_f32_16x16x32_f16(false, a, false, b, (short)0, c, false, false);
}

// ---------------------------------------------------------------- converts
__global__ void k_f32_to_f16(const float* __restrict__ in, _Float16* __restrict__ out, int n) {
  for (int i = blockIdx.x * blockDim.x + threadIdx.x; i < n; i += gridDim.x * blockDim.x)
    out[i] = (_Float16)in[i];
}

// ---------------------------------------------------------------- GEMM: C[M,N] = A[M,K] * Bw[N,K]^T + bias
// Block: 128 threads = 4 waves. Each wave register-blocks a 32(M)x64(N) tile
// (2 m-subtiles x 4 n-subtiles = 8 WMMA accumulators). The 64x32-half B tile
// for each K-step is staged once into LDS (async-to-LDS when available) and
// shared by all 4 waves.
template<bool OUT_F16>
__global__ __launch_bounds__(128)
void k_gemm(const _Float16* __restrict__ A, const _Float16* __restrict__ Bw,
            const float* __restrict__ bias, void* __restrict__ Cout,
            int Ndim, int Kdim)
{
  __shared__ __align__(32) _Float16 Blds[64 * 32];   // 4 KB: 64 N-rows x 32 K-halves
  const int tid  = threadIdx.x;
  const int lane = tid & 31;
  const int wave = tid >> 5;                 // 0..3 -> M sub-block
  const int half = lane >> 4;
  const int l16  = lane & 15;
  const int c0 = half * 16;   // A fragment: halves [c0,c0+16) of a 32-K chunk
  const int c1 = half * 16;   // B fragment: halves [c1,c1+16) of a 32-K chunk
  const int mbase = blockIdx.x * 128 + wave * 32;
  const int nbase = blockIdx.y * 64;

  const _Float16* arow0 = A + (size_t)(mbase + l16) * Kdim;
  const _Float16* arow1 = arow0 + (size_t)16 * Kdim;

  v8f acc[2][4];
#pragma unroll
  for (int i = 0; i < 2; ++i)
#pragma unroll
    for (int j = 0; j < 4; ++j) acc[i][j] = (v8f){};

  for (int kk = 0; kk < Kdim; kk += 32) {
    // ---- stage the shared B tile (64 rows x 32 halves) for this K-chunk
#pragma unroll
    for (int u = 0; u < 2; ++u) {
      const int op  = tid + u * 128;          // 0..255 b128 transfers
      const int row = op >> 2;
      const int ch  = (op & 3) * 8;
      const _Float16* src = Bw + (size_t)(nbase + row) * Kdim + kk + ch;
      _Float16* dst = &Blds[row * 32 + ch];
#if HAS_ASYNC_LDS
      __builtin_amdgcn_global_load_async_to_lds_b128(
          (__attribute__((address_space(1))) v4i*)src,
          (__attribute__((address_space(3))) v4i*)dst, 0, 0);
#else
      *(v8h*)dst = *(const v8h*)src;
#endif
    }
#if HAS_ASYNC_LDS
#if __has_builtin(__builtin_amdgcn_s_wait_asynccnt)
    __builtin_amdgcn_s_wait_asynccnt(0);
#else
    asm volatile("s_wait_asynccnt 0x0" ::: "memory");
#endif
#endif
    __syncthreads();

    const v16h a0 = cat8(*(const v8h*)(arow0 + kk + c0), *(const v8h*)(arow0 + kk + c0 + 8));
    const v16h a1 = cat8(*(const v8h*)(arow1 + kk + c0), *(const v8h*)(arow1 + kk + c0 + 8));
#pragma unroll
    for (int j = 0; j < 4; ++j) {
      const v16h bf = *(const v16h*)(&Blds[(j * 16 + l16) * 32 + c1]);
      acc[0][j] = wmma_f16(a0, bf, acc[0][j]);
      acc[1][j] = wmma_f16(a1, bf, acc[1][j]);
    }
    __syncthreads();
  }

  // C/D layout: lanes 0-15 -> rows 0..7, lanes 16-31 -> rows 8..15; col = lane%16
  const int r0 = half * 8;
#pragma unroll
  for (int i = 0; i < 2; ++i) {
#pragma unroll
    for (int j = 0; j < 4; ++j) {
      const int ccol = nbase + j * 16 + l16;
      const float bz = bias[ccol];
#pragma unroll
      for (int r = 0; r < 8; ++r) {
        const size_t idx = (size_t)(mbase + i * 16 + r0 + r) * Ndim + ccol;
        const float v = acc[i][j][r] + bz;
        if (OUT_F16) ((_Float16*)Cout)[idx] = (_Float16)v;
        else         ((float*)Cout)[idx]    = v;
      }
    }
  }
}

// ---------------------------------------------------------------- 4D RoPE in-place on f16 [tok, 1024]
// one thread per rotation pair -> race-free in-place
__global__ void k_rope(_Float16* __restrict__ qk, const float* __restrict__ coords,
                       const float* __restrict__ f0, const float* __restrict__ f1,
                       const float* __restrict__ f2, int ntok)
{
  const float HPI = 1.5707963267948966f;
  const int total = ntok * (DDim / 2);
  for (int i = blockIdx.x * blockDim.x + threadIdx.x; i < total; i += gridDim.x * blockDim.x) {
    const int p   = i & 511;   // pair index within the 1024-wide row
    const int tok = i >> 9;
    const int d2  = p & 31;    // pair index within head (DH/2 = 32)
    int axis; float fr;
    if      (d2 < 12) { axis = 0; fr = f0[d2];      }
    else if (d2 < 22) { axis = 1; fr = f1[d2 - 12]; }
    else              { axis = 2; fr = f2[d2 - 22]; }
    const float arg = HPI * coords[tok * 3 + axis] * fr;
    float s, c; __sincosf(arg, &s, &c);
    const size_t e = (size_t)tok * DDim + 2 * p;
    const float x1 = (float)qk[e], x2 = (float)qk[e + 1];
    qk[e]     = (_Float16)(x1 * c - x2 * s);
    qk[e + 1] = (_Float16)(x2 * c + x1 * s);
  }
}

// ---------------------------------------------------------------- V: [B,T,NK,D] -> [B,T,H,DH,NK] (dh-major for WMMA B fragments)
__global__ void k_vt(const _Float16* __restrict__ V, _Float16* __restrict__ Vt, int n)
{
  for (int i = blockIdx.x * blockDim.x + threadIdx.x; i < n; i += gridDim.x * blockDim.x) {
    const int o  = i % DDim;
    const int nr = (i / DDim) % NKd;
    const int bt = i / (DDim * NKd);
    const int h  = o >> 6, dh = o & 63;
    Vt[((size_t)bt * HH + h) * (DH * NKd) + (size_t)dh * NKd + nr] = V[i];
  }
}

// ---------------------------------------------------------------- attention: one wave per (b,h,t, 16-row q-tile)
__global__ __launch_bounds__(32)
void k_attn(const _Float16* __restrict__ Q, const _Float16* __restrict__ K,
            const _Float16* __restrict__ Vt,
            const float* __restrict__ cq, const float* __restrict__ ck,
            const unsigned char* __restrict__ qpad, const unsigned char* __restrict__ kpad,
            const float* __restrict__ alpha, _Float16* __restrict__ Y)
{
  __shared__ __align__(32) float    sbuf[16 * NKd];   // scores (f32)
  __shared__ __align__(32) _Float16 pbuf[16 * NKd];   // exp(s-m) (f16, A-fragment friendly)
  __shared__ float lbuf[16];

  const int lane = threadIdx.x;
  const int qt   = blockIdx.x;                // 0..35
  const int bht  = blockIdx.y;                // b*H*T + h*T + t
  const int t = bht % TT;
  const int h = (bht / TT) % HH;
  const int b = bht / (TT * HH);
  const int bt = b * TT + t;
  const int half = lane >> 4, l16 = lane & 15;
  const int c0 = half * 16, c1 = half * 16, r0 = half * 8;

  const size_t qbase = ((size_t)bt * NQd) * DDim + h * DH;
  const size_t kbase = ((size_t)bt * NKd) * DDim + h * DH;
  const size_t vbase = ((size_t)bt * HH + h) * (size_t)(DH * NKd);

  // Q A-fragments (K = 64 split into two 32-chunks), constant over the k-tile loop
  const _Float16* qrow = Q + qbase + (size_t)(qt * 16 + l16) * DDim;
  const v16h aq0 = cat8(*(const v8h*)(qrow + c0),      *(const v8h*)(qrow + c0 + 8));
  const v16h aq1 = cat8(*(const v8h*)(qrow + 32 + c0), *(const v8h*)(qrow + 32 + c0 + 8));

  // per-output-row (epilogue rows r0..r0+7) bias data
  float qy[8], qx[8]; bool qm[8];
#pragma unroll
  for (int r = 0; r < 8; ++r) {
    const int qi = qt * 16 + r0 + r;
    qy[r] = cq[((size_t)bt * NQd + qi) * 3 + 1];
    qx[r] = cq[((size_t)bt * NQd + qi) * 3 + 2];
    qm[r] = qpad[(size_t)bt * NQd + qi] != 0;
  }
  const float al = alpha[h];

  // ---- scores: S = (Q K^T)/8 + bias, staged to LDS
  for (int kt = 0; kt < NKd / 16; ++kt) {
    const int ki = kt * 16 + l16;
    const _Float16* krow = K + kbase + (size_t)ki * DDim;
    const v16h bk0 = *(const v16h*)(krow + c1);
    const v16h bk1 = *(const v16h*)(krow + 32 + c1);
    v8f s = {};
    s = wmma_f16(aq0, bk0, s);
    s = wmma_f16(aq1, bk1, s);
    const float ky = ck[((size_t)bt * NKd + ki) * 3 + 1];
    const float kx = ck[((size_t)bt * NKd + ki) * 3 + 2];
    const bool  km = kpad[(size_t)bt * NKd + ki] != 0;
#pragma unroll
    for (int r = 0; r < 8; ++r) {
      const float dy = qy[r] - ky, dx = qx[r] - kx;
      const float dist = __builtin_amdgcn_sqrtf(dy * dy + dx * dx);   // raw v_sqrt_f32
      float bias = ((dist > 0.5f) ? NEGBIG : 0.0f) + al * __expf(-10.0f * dist);
      if (qm[r] | km) bias = NEGBIG;
      sbuf[(r0 + r) * NKd + ki] = s[r] * 0.125f + bias;
    }
  }
  __syncthreads();

  // ---- row softmax (lane pair {l16, l16+16} shares row l16; each does half the columns)
  {
    const int row = l16;
    const int cb = half * (NKd / 2);
    float m = -3.4e38f;
    for (int c2 = 0; c2 < NKd / 2; ++c2) m = fmaxf(m, sbuf[row * NKd + cb + c2]);
    m = fmaxf(m, __shfl_xor(m, 16, 32));
    float lsum = 0.0f;
    for (int c2 = 0; c2 < NKd / 2; ++c2) {
      const float p = __expf(sbuf[row * NKd + cb + c2] - m);
      lsum += p;
      pbuf[row * NKd + cb + c2] = (_Float16)p;   // unnormalized; 1/l folded into O
    }
    lsum += __shfl_xor(lsum, 16, 32);
    if (lane < 16) lbuf[row] = 1.0f / lsum;
  }
  __syncthreads();

  // ---- O = P @ V  (4 column tiles of 16 covering DH=64, K-loop over 576 keys)
  v8f o0 = {}, o1 = {}, o2 = {}, o3 = {};
  const _Float16* prow = &pbuf[l16 * NKd];
  for (int kk = 0; kk < NKd; kk += 32) {
    const v16h ap = cat8(*(const v8h*)(prow + kk + c0), *(const v8h*)(prow + kk + c0 + 8));
    const _Float16* vb = Vt + vbase + (size_t)l16 * NKd + kk + c1;
    o0 = wmma_f16(ap, *(const v16h*)(vb + 0 * 16 * NKd), o0);
    o1 = wmma_f16(ap, *(const v16h*)(vb + 1 * 16 * NKd), o1);
    o2 = wmma_f16(ap, *(const v16h*)(vb + 2 * 16 * NKd), o2);
    o3 = wmma_f16(ap, *(const v16h*)(vb + 3 * 16 * NKd), o3);
  }
#pragma unroll
  for (int r = 0; r < 8; ++r) {
    const int qi = qt * 16 + r0 + r;
    const float sc = lbuf[r0 + r];
    const size_t base = ((size_t)bt * NQd + qi) * DDim + h * DH + l16;
    Y[base +  0] = (_Float16)(o0[r] * sc);
    Y[base + 16] = (_Float16)(o1[r] * sc);
    Y[base + 32] = (_Float16)(o2[r] * sc);
    Y[base + 48] = (_Float16)(o3[r] * sc);
  }
}

// ---------------------------------------------------------------- host
extern "C" void kernel_launch(void* const* d_in, const int* in_sizes, int n_in,
                              void* d_out, int out_size, void* d_ws, size_t ws_size,
                              hipStream_t stream) {
  (void)in_sizes; (void)n_in; (void)out_size; (void)ws_size;
  const float* q_tokens  = (const float*)d_in[0];
  const float* kv_tokens = (const float*)d_in[1];
  const float* coords_q  = (const float*)d_in[2];
  const float* coords_k  = (const float*)d_in[3];
  const unsigned char* q_pad  = (const unsigned char*)d_in[4];   // jnp bool_ = 1 byte
  const unsigned char* kv_pad = (const unsigned char*)d_in[5];
  const float* Wq = (const float*)d_in[6];  const float* bq = (const float*)d_in[7];
  const float* Wk = (const float*)d_in[8];  const float* bk = (const float*)d_in[9];
  const float* Wv = (const float*)d_in[10]; const float* bv = (const float*)d_in[11];
  const float* Wo = (const float*)d_in[12]; const float* bo = (const float*)d_in[13];
  const float* alpha = (const float*)d_in[14];
  const float* f0 = (const float*)d_in[15];
  const float* f1 = (const float*)d_in[16];
  const float* f2 = (const float*)d_in[17];

  // workspace layout (all f16)
  char* ws = (char*)d_ws;
  const size_t sz_tok = (size_t)MTOK * DDim * sizeof(_Float16);   // 9.44 MB
  const size_t sz_w   = (size_t)DDim * DDim * sizeof(_Float16);   // 2.10 MB
  _Float16* Xq  = (_Float16*)ws; ws += sz_tok;
  _Float16* Xkv = (_Float16*)ws; ws += sz_tok;
  _Float16* Wqh = (_Float16*)ws; ws += sz_w;
  _Float16* Wkh = (_Float16*)ws; ws += sz_w;
  _Float16* Wvh = (_Float16*)ws; ws += sz_w;
  _Float16* Woh = (_Float16*)ws; ws += sz_w;
  _Float16* Qp  = (_Float16*)ws; ws += sz_tok;   // Q projection (then RoPE'd in place)
  _Float16* Kp  = (_Float16*)ws; ws += sz_tok;
  _Float16* Vp  = (_Float16*)ws; ws += sz_tok;
  _Float16* Vt  = (_Float16*)ws; ws += sz_tok;   // [B,T,H,DH,NK]
  _Float16* Yh  = (_Float16*)ws; ws += sz_tok;   // attention output [B,T,NQ,D]

  const int cthr = 256;
  const int nTok = MTOK * DDim;    // 4,718,592
  const int nW   = DDim * DDim;    // 1,048,576

  // 1) f32 -> f16 conversions
  k_f32_to_f16<<<(nTok + cthr - 1) / cthr, cthr, 0, stream>>>(q_tokens,  Xq,  nTok);
  k_f32_to_f16<<<(nTok + cthr - 1) / cthr, cthr, 0, stream>>>(kv_tokens, Xkv, nTok);
  k_f32_to_f16<<<(nW + cthr - 1) / cthr, cthr, 0, stream>>>(Wq, Wqh, nW);
  k_f32_to_f16<<<(nW + cthr - 1) / cthr, cthr, 0, stream>>>(Wk, Wkh, nW);
  k_f32_to_f16<<<(nW + cthr - 1) / cthr, cthr, 0, stream>>>(Wv, Wvh, nW);
  k_f32_to_f16<<<(nW + cthr - 1) / cthr, cthr, 0, stream>>>(Wo, Woh, nW);

  // 2) projections: [4608,1024] x [1024,1024]^T ; grid = (M/128, N/64), 128 threads
  dim3 ggrid(MTOK / 128, DDim / 64);
  k_gemm<true><<<ggrid, 128, 0, stream>>>(Xq,  Wqh, bq, Qp, DDim, DDim);
  k_gemm<true><<<ggrid, 128, 0, stream>>>(Xkv, Wkh, bk, Kp, DDim, DDim);
  k_gemm<true><<<ggrid, 128, 0, stream>>>(Xkv, Wvh, bv, Vp, DDim, DDim);

  // 3) RoPE in-place on Q and K
  const int nPairQ = MTOK * (DDim / 2);
  k_rope<<<(nPairQ + cthr - 1) / cthr, cthr, 0, stream>>>(Qp, coords_q, f0, f1, f2, MTOK);
  k_rope<<<(nPairQ + cthr - 1) / cthr, cthr, 0, stream>>>(Kp, coords_k, f0, f1, f2, MTOK);

  // 4) V transpose to dh-major
  k_vt<<<(nTok + cthr - 1) / cthr, cthr, 0, stream>>>(Vp, Vt, nTok);

  // 5) attention: one wave per (b,h,t, q-tile)
  dim3 agrid(NQd / 16, BB * HH * TT);
  k_attn<<<agrid, 32, 0, stream>>>(Qp, Kp, Vt, coords_q, coords_k, q_pad, kv_pad, alpha, Yh);

  // 6) output projection -> f32 d_out
  k_gemm<false><<<ggrid, 128, 0, stream>>>(Yh, Woh, bo, d_out, DDim, DDim);
}
